// CustomLSTM_77799037599829
// MI455X (gfx1250) — matrix operand order repaired and checked by reading, not compiled, not run on hardware
//
#include <hip/hip_runtime.h>
#include <math.h>

// ---------------- problem constants ----------------
#define BB   64
#define SS   512
#define HH   1024
#define IIN  1024
#define G4   4096            // 4*H
#define MROWS (SS*BB)        // 32768 GEMM rows (row index = s*B + b)

#define SCALEC 1.0f
#define SHIFTC 0.0f

// scan kernel geometry: 32 blocks x 256 threads = 256 waves; wave tile 64(M) x 16(N)
#define SCAN_BLOCKS 32

// ---------------- workspace layout (bytes) ----------------
static constexpr size_t OFF_XP     = 0;                                       // float [MROWS][4096]
static constexpr size_t OFF_XBF    = OFF_XP    + (size_t)MROWS * G4 * 4;      // ushort [MROWS][1024]
static constexpr size_t OFF_OUT0BF = OFF_XBF   + (size_t)MROWS * IIN * 2;     // ushort [MROWS][1024]
static constexpr size_t OFF_W      = OFF_OUT0BF+ (size_t)MROWS * HH * 2;      // 4 x ushort [4096][1024]
static constexpr size_t OFF_GATES  = OFF_W     + 4ull * G4 * HH * 2;          // float [64][4096]
static constexpr size_t OFF_C      = OFF_GATES + (size_t)BB * G4 * 4;         // float [64][1024]
static constexpr size_t OFF_HBF    = OFF_C     + (size_t)BB * HH * 4;         // ushort [64][1024]
static constexpr size_t OFF_BAR    = OFF_HBF   + (size_t)BB * HH * 2;         // 2 x uint

// d_out layout: out1 (B,S,H) | h_n (2,B,H) | c_n (2,B,H)
static constexpr size_t OUT1_N  = (size_t)BB * SS * HH;        // 33554432
static constexpr size_t HN_OFF  = OUT1_N;
static constexpr size_t CN_OFF  = OUT1_N + 2ull * BB * HH;

// ---------------- WMMA types / fragments ----------------
typedef __attribute__((ext_vector_type(16))) __bf16 v16bf;
typedef __attribute__((ext_vector_type(8)))  float  v8f;

union FragU { v16bf v; uint4 u[2]; };

// fp32 -> bf16 round-to-nearest-even, stored as raw u16
__device__ __forceinline__ unsigned short f2bf(float f) {
  unsigned u = __float_as_uint(f);
  unsigned r = (u >> 16) & 1u;
  u += 0x7fffu + r;
  return (unsigned short)(u >> 16);
}

// A fragment: 16x32 bf16 tile at (row0,k0); base points at element (row0,k0).
// ISA layout: lane<16 -> row=lane, K = {0..7, 16..23}; lane>=16 -> row=lane-16, K = {8..15, 24..31}
__device__ __forceinline__ v16bf load_frag_a(const unsigned short* __restrict__ base,
                                             int pitch, int lane) {
  const int r  = lane & 15;
  const int kh = lane >> 4;
  const unsigned short* p = base + (size_t)r * pitch;
  FragU f;
  f.u[0] = *reinterpret_cast<const uint4*>(p + kh * 8);
  f.u[1] = *reinterpret_cast<const uint4*>(p + 16 + kh * 8);
  return f.v;
}

// B fragment: 32x16 bf16 tile of W^T, W stored row-major [out(N)][in(K)].
// wbase points at W element (n0, k0). Column n of B == row n of W; K = {8kh..8kh+7, 16+8kh..16+8kh+7} per lane.
__device__ __forceinline__ v16bf load_frag_b(const unsigned short* __restrict__ wbase,
                                             int pitch, int lane) {
  const int n  = lane & 15;
  const int kh = lane >> 4;
  const unsigned short* p = wbase + (size_t)n * pitch + kh * 8;
  FragU f;
  f.u[0] = *reinterpret_cast<const uint4*>(p);
  f.u[1] = *reinterpret_cast<const uint4*>(p + 16);
  return f.v;
}

__device__ __forceinline__ v8f wmma_bf16(v16bf a, v16bf b, v8f c) {
  return __builtin_amdgcn_wmma_f32_16x16x32_bf16(false, a, false, b, (short)0, c, false, false);
}

__device__ __forceinline__ float sigmoidf_s(float x) {
  return SCALEC * (1.0f / (1.0f + __expf(-x))) + SHIFTC;
}

// ---- CDNA5 async global->LDS copy (tracked by ASYNCcnt) ----
__device__ __forceinline__ void async_g2l_b128(unsigned lds_byte, const void* gaddr) {
  asm volatile("global_load_async_to_lds_b128 %0, %1, off"
               :: "v"(lds_byte), "v"(gaddr) : "memory");
}
__device__ __forceinline__ void wait_async0() {
  asm volatile("s_wait_asynccnt 0x0" ::: "memory");
}

// ---------------- small prep kernels ----------------
__global__ void k_f2bf(unsigned short* __restrict__ dst, const float* __restrict__ src, int n) {
  for (int i = blockIdx.x * blockDim.x + threadIdx.x; i < n; i += gridDim.x * blockDim.x)
    dst[i] = f2bf(src[i]);
}

// x: (B,S,IN) fp32 -> xbf rows = s*B+b, bf16
__global__ void k_pack_x(unsigned short* __restrict__ dst, const float* __restrict__ x) {
  const int n = MROWS * IIN;
  for (int i = blockIdx.x * blockDim.x + threadIdx.x; i < n; i += gridDim.x * blockDim.x) {
    int k   = i & (IIN - 1);
    int row = i >> 10;            // s*B + b
    int b   = row & (BB - 1);
    int s   = row >> 6;
    dst[i] = f2bf(x[((size_t)b * SS + s) * IIN + k]);
  }
}

__global__ void k_init_state(float* __restrict__ c, unsigned short* __restrict__ hbf,
                             unsigned* __restrict__ bar) {
  int i = blockIdx.x * blockDim.x + threadIdx.x;
  if (i < BB * HH) { c[i] = 0.0f; hbf[i] = 0; }
  if (i < 2) bar[i] = 0u;
}

// ---------------- big input-projection GEMM ----------------
// XP[m][n] = sum_k A[m][k] * W[n][k] + bias[n]   (A: MROWS x 1024 bf16, W: 4096 x 1024 bf16)
// block tile 128(M) x 128(N); 8 waves, wave tile 32x64.
__global__ __launch_bounds__(256) void k_gemm_xp(float* __restrict__ XP,
                                                 const unsigned short* __restrict__ A,
                                                 const unsigned short* __restrict__ W,
                                                 const float* __restrict__ bias) {
  const int lane = threadIdx.x & 31;
  const int wid  = threadIdx.x >> 5;       // 0..7
  const int wm   = wid & 3;                // 4 M slices of 32
  const int wn   = wid >> 2;               // 2 N slices of 64
  const int m0   = blockIdx.y * 128 + wm * 32;
  const int n0   = blockIdx.x * 128 + wn * 64;

  v8f acc[2][4] = {};
  for (int k = 0; k < IIN; k += 32) {
    v16bf a0 = load_frag_a(A + (size_t)m0 * IIN + k, IIN, lane);
    v16bf a1 = load_frag_a(A + (size_t)(m0 + 16) * IIN + k, IIN, lane);
#pragma unroll
    for (int j = 0; j < 4; ++j) {
      v16bf b = load_frag_b(W + (size_t)(n0 + j * 16) * IIN + k, IIN, lane);
      acc[0][j] = wmma_bf16(a0, b, acc[0][j]);
      acc[1][j] = wmma_bf16(a1, b, acc[1][j]);
    }
  }
  const int cn   = lane & 15;
  const int half = lane >> 4;
#pragma unroll
  for (int i = 0; i < 2; ++i) {
    const int mrow = m0 + i * 16 + half * 8;
#pragma unroll
    for (int j = 0; j < 4; ++j) {
      const int col = n0 + j * 16 + cn;
      const float bv = bias[col];
      float* p = XP + (size_t)mrow * G4 + col;
#pragma unroll
      for (int r = 0; r < 8; ++r) p[(size_t)r * G4] = acc[i][j][r] + bv;
    }
  }
}

// ---------------- device-wide barrier ----------------
__device__ __forceinline__ void grid_barrier(unsigned* __restrict__ bar, unsigned nblocks) {
  __syncthreads();
  if (threadIdx.x == 0) {
    __threadfence();
    unsigned g = atomicAdd(&bar[1], 0u);
    if (atomicAdd(&bar[0], 1u) == nblocks - 1u) {
      bar[0] = 0u;
      __threadfence();
      atomicAdd(&bar[1], 1u);
    } else {
      while (atomicAdd(&bar[1], 0u) == g) { __builtin_amdgcn_s_sleep(2); }
    }
    __threadfence();
  }
  __syncthreads();
}

// ---------------- persistent recurrent scan ----------------
// 32 blocks x 256 threads = 256 waves, wave tile 64(M) x 16(N): zero W_hh redundancy.
// Per step: async-stage h (64x1024 bf16, 128KB) into LDS; gates = h @ W_hh^T via WMMA
// (A-frags from LDS, B from global/L2); add XP + b_hh; barrier; elementwise cell; barrier.
__global__ __launch_bounds__(256) void k_scan(float* __restrict__ gates,
                                              const float* __restrict__ XP,
                                              const unsigned short* __restrict__ Whh,
                                              const float* __restrict__ bhh,
                                              float* __restrict__ cbuf,
                                              unsigned short* __restrict__ hbf,
                                              unsigned short* __restrict__ outbf,  // layer0: next-layer bf16 input
                                              float* __restrict__ dout,            // out1 + h_n/c_n tails
                                              unsigned* __restrict__ bar,
                                              int layer) {
  __shared__ unsigned short lds_h[BB * HH];   // 128 KB (WGP has 320 KB)

  const int lane = threadIdx.x & 31;
  const int wid  = blockIdx.x * 8 + (threadIdx.x >> 5);   // 0..255
  const int n0   = wid * 16;                              // 16 gate columns per wave
  const int tid  = blockIdx.x * 256 + threadIdx.x;        // 0..8191
  const int cn   = lane & 15;
  const int half = lane >> 4;

  const unsigned lds_base = (unsigned)(uintptr_t)&lds_h[0];
  const char*    hsrc     = (const char*)hbf;

  for (int s = 0; s < SS; ++s) {
    // ---- stage h into LDS via CDNA5 async copy (128KB: 256 thr x 32 x 16B) ----
#pragma unroll 4
    for (int i = 0; i < 32; ++i) {
      unsigned off = (unsigned)threadIdx.x * 16u + (unsigned)i * 4096u;
      async_g2l_b128(lds_base + off, hsrc + off);
    }
    wait_async0();
    __syncthreads();

    // ---- phase 1: gate pre-activations via WMMA ----
    v8f acc[4] = {};
    for (int k = 0; k < HH; k += 32) {
      v16bf b = load_frag_b(Whh + (size_t)n0 * HH + k, HH, lane);
#pragma unroll
      for (int mc = 0; mc < 4; ++mc) {
        v16bf a = load_frag_a(lds_h + (size_t)(mc * 16) * HH + k, HH, lane);
        acc[mc] = wmma_bf16(a, b, acc[mc]);
      }
    }
    {
      const int col = n0 + cn;
      const float bv = bhh[col];
#pragma unroll
      for (int mc = 0; mc < 4; ++mc) {
        const int mrow = mc * 16 + half * 8;
        const float* xp = XP + ((size_t)s * BB + mrow) * G4 + col;
        float* gp = gates + (size_t)mrow * G4 + col;
#pragma unroll
        for (int r = 0; r < 8; ++r)
          gp[(size_t)r * G4] = acc[mc][r] + xp[(size_t)r * G4] + bv;
      }
    }
    grid_barrier(bar, gridDim.x);

    // ---- phase 2: elementwise LSTM cell ----
    for (int idx = tid; idx < BB * HH; idx += SCAN_BLOCKS * 256) {
      const int b = idx >> 10;
      const int j = idx & (HH - 1);
      const float* g = gates + (size_t)b * G4;
      const float ig = sigmoidf_s(g[j]);
      const float fg = sigmoidf_s(g[HH + j]);
      const float gg = tanhf(g[2 * HH + j]);
      const float og = sigmoidf_s(g[3 * HH + j]);
      const float cprev = cbuf[idx];
      const float cnew  = fg * cprev + ig * gg;
      const float hnew  = og * tanhf(cnew);
      cbuf[idx] = cnew;
      hbf[idx]  = f2bf(hnew);
      if (outbf) {
        outbf[((size_t)s * BB + b) * HH + j] = f2bf(hnew);      // layer0 -> layer1 input
      } else {
        dout[((size_t)b * SS + s) * HH + j] = hnew;             // layer1 -> out1
      }
      if (s == SS - 1) {
        dout[HN_OFF + (size_t)layer * BB * HH + idx] = hnew;
        dout[CN_OFF + (size_t)layer * BB * HH + idx] = cnew;
      }
    }
    grid_barrier(bar, gridDim.x);
  }
}

// ---------------- launch ----------------
extern "C" void kernel_launch(void* const* d_in, const int* in_sizes, int n_in,
                              void* d_out, int out_size, void* d_ws, size_t ws_size,
                              hipStream_t stream) {
  const float* x     = (const float*)d_in[0];
  const float* W_ih0 = (const float*)d_in[1];
  const float* b_ih0 = (const float*)d_in[2];
  const float* W_hh0 = (const float*)d_in[3];
  const float* b_hh0 = (const float*)d_in[4];
  const float* W_ih1 = (const float*)d_in[5];
  const float* b_ih1 = (const float*)d_in[6];
  const float* W_hh1 = (const float*)d_in[7];
  const float* b_hh1 = (const float*)d_in[8];
  float* out = (float*)d_out;
  char*  ws  = (char*)d_ws;

  float*          XP     = (float*)(ws + OFF_XP);
  unsigned short* xbf    = (unsigned short*)(ws + OFF_XBF);
  unsigned short* out0bf = (unsigned short*)(ws + OFF_OUT0BF);
  unsigned short* wih0   = (unsigned short*)(ws + OFF_W);
  unsigned short* whh0   = wih0 + (size_t)G4 * HH;
  unsigned short* wih1   = whh0 + (size_t)G4 * HH;
  unsigned short* whh1   = wih1 + (size_t)G4 * HH;
  float*          gates  = (float*)(ws + OFF_GATES);
  float*          cbuf   = (float*)(ws + OFF_C);
  unsigned short* hbf    = (unsigned short*)(ws + OFF_HBF);
  unsigned*       bar    = (unsigned*)(ws + OFF_BAR);

  const int WN = G4 * HH;  // 4194304 elements per weight matrix

  // weight + input conversion to bf16
  k_f2bf<<<1024, 256, 0, stream>>>(wih0, W_ih0, WN);
  k_f2bf<<<1024, 256, 0, stream>>>(whh0, W_hh0, WN);
  k_f2bf<<<1024, 256, 0, stream>>>(wih1, W_ih1, WN);
  k_f2bf<<<1024, 256, 0, stream>>>(whh1, W_hh1, WN);
  k_pack_x<<<2048, 256, 0, stream>>>(xbf, x);

  dim3 ggrid(G4 / 128, MROWS / 128);  // (32, 256)

  // ---- layer 0 ----
  k_init_state<<<256, 256, 0, stream>>>(cbuf, hbf, bar);
  k_gemm_xp<<<ggrid, 256, 0, stream>>>(XP, xbf, wih0, b_ih0);
  k_scan<<<SCAN_BLOCKS, 256, 0, stream>>>(gates, XP, whh0, b_hh0, cbuf, hbf, out0bf, out, bar, 0);

  // ---- layer 1 ----
  k_init_state<<<256, 256, 0, stream>>>(cbuf, hbf, bar);
  k_gemm_xp<<<ggrid, 256, 0, stream>>>(XP, out0bf, wih1, b_ih1);
  k_scan<<<SCAN_BLOCKS, 256, 0, stream>>>(gates, XP, whh1, b_hh1, cbuf, hbf, nullptr, out, bar, 1);
}
